// attn_head_76459007803804
// MI455X (gfx1250) — matrix operand order repaired and checked by reading, not compiled, not run on hardware
//
#include <hip/hip_runtime.h>
#include <hip/hip_bf16.h>
#include <math.h>

#define N_ROWS 8192
#define F_IN   512
#define F_OUT  128

typedef __attribute__((ext_vector_type(16))) _Float16 v16h;
typedef __attribute__((ext_vector_type(8)))  float    v8f;

typedef __attribute__((ext_vector_type(4))) unsigned int u32x4;
typedef __attribute__((ext_vector_type(8))) int          i32x8;
typedef __attribute__((ext_vector_type(4))) int          i32x4;

#if __has_builtin(__builtin_amdgcn_tensor_load_to_lds)
#define HAVE_TDM 1
#else
#define HAVE_TDM 0
#endif

// D = A(16x32 f16) * B(32x16 f16) + C(16x16 f32)
__device__ __forceinline__ v8f wmma16(v16h a, v16h b, v8f c) {
  return __builtin_amdgcn_wmma_f32_16x16x32_f16(false, a, false, b,
                                                (short)0, c, false, false);
}

// branch-free tanh = 1 - 2/(e^2x + 1); hardware v_exp + v_rcp + fma
__device__ __forceinline__ float fast_tanh(float x) {
  float cx = fminf(fmaxf(x, -9.0f), 9.0f);
  float e  = __expf(2.0f * cx);
  return 1.0f - 2.0f * __builtin_amdgcn_rcpf(e + 1.0f);
}

// ---- fragment loaders (wave32, contiguous k order, same for A and B) ----

// A 16x32: element (m,k): m = lane%16, k = 16*(lane/16) + h
__device__ __forceinline__ v16h load_A_f32(const float* __restrict__ src, int ld,
                                           int r0, int c0) {
  const int lane = threadIdx.x & 31;
  const int m = lane & 15;
  const int kb = (lane >> 4) << 4;
  v16h a;
#pragma unroll
  for (int h = 0; h < 16; ++h) {
    int k = kb + h;
    a[h] = (_Float16)src[(size_t)(r0 + m) * ld + c0 + k];
  }
  return a;
}

__device__ __forceinline__ v16h load_A_f16(const _Float16* __restrict__ src, int ld,
                                           int r0, int c0) {
  const int lane = threadIdx.x & 31;
  const int m = lane & 15;
  const int kb = (lane >> 4) << 4;
  v16h a;
#pragma unroll
  for (int h = 0; h < 16; ++h) {
    int k = kb + h;
    a[h] = src[(size_t)(r0 + m) * ld + c0 + k];
  }
  return a;
}

// A 16x32 from a row-major [16][32] f16 LDS tile
__device__ __forceinline__ v16h load_A_lds(const _Float16* t) {
  const int lane = threadIdx.x & 31;
  const int m = lane & 15;
  const int kb = (lane >> 4) << 4;
  v16h a;
#pragma unroll
  for (int h = 0; h < 16; ++h) {
    int k = kb + h;
    a[h] = t[m * 32 + k];
  }
  return a;
}

// A 16x32 from a row-major [16][32] f32 LDS tile (convert to f16)
__device__ __forceinline__ v16h load_A_lds_f32(const float* t) {
  const int lane = threadIdx.x & 31;
  const int m = lane & 15;
  const int kb = (lane >> 4) << 4;
  v16h a;
#pragma unroll
  for (int h = 0; h < 16; ++h) {
    int k = kb + h;
    a[h] = (_Float16)t[m * 32 + k];
  }
  return a;
}

// B 32x16: element (k,n): n = lane%16, k = h + 16*(lane/16)
// "transposed" source: B(k,n) = src[(n0+n)*ld + k0+k]
__device__ __forceinline__ v16h load_Bt_f32(const float* __restrict__ src, int ld,
                                            int n0, int k0) {
  const int lane = threadIdx.x & 31;
  const int n = lane & 15;
  const int kb = (lane >> 4) << 4;
  v16h b;
#pragma unroll
  for (int h = 0; h < 16; ++h)
    b[h] = (_Float16)src[(size_t)(n0 + n) * ld + k0 + kb + h];
  return b;
}

// same, from a row-major [32][128] f16 LDS tile
__device__ __forceinline__ v16h load_Bt_lds(const _Float16* t, int n0, int k0) {
  const int lane = threadIdx.x & 31;
  const int n = lane & 15;
  const int kb = (lane >> 4) << 4;
  v16h b;
#pragma unroll
  for (int h = 0; h < 16; ++h)
    b[h] = t[(n0 + n) * F_OUT + k0 + kb + h];
  return b;
}

// row-major [K][N] source: B(k,n) = src[(k0+k)*ld + n0+n]
__device__ __forceinline__ v16h load_B_rm_f16(const _Float16* __restrict__ src, int ld,
                                              int k0, int n0) {
  const int lane = threadIdx.x & 31;
  const int n = lane & 15;
  const int kb = (lane >> 4) << 4;
  v16h b;
#pragma unroll
  for (int h = 0; h < 16; ++h)
    b[h] = src[(size_t)(k0 + kb + h) * ld + n0 + n];
  return b;
}

// same, from the row-major [32][128] f16 LDS tile (k = tile row)
__device__ __forceinline__ v16h load_B_rm_lds(const _Float16* t, int n0) {
  const int lane = threadIdx.x & 31;
  const int n = lane & 15;
  const int kb = (lane >> 4) << 4;
  v16h b;
#pragma unroll
  for (int h = 0; h < 16; ++h)
    b[h] = t[(kb + h) * F_OUT + n0 + n];
  return b;
}

#if HAVE_TDM
// Issue a TDM 2D tile load into LDS. D# bit layout per CDNA5 ISA ch.8.
// dsize_code: 0=1B, 1=2B, 2=4B elements. Dims/strides in elements.
__device__ __forceinline__ void tdm_load_2d(const void* gsrc, unsigned lds_off,
                                            unsigned dsize_code,
                                            unsigned td0, unsigned td1,
                                            unsigned tl0, unsigned tl1,
                                            unsigned stride0) {
  unsigned long long ga = (unsigned long long)(size_t)gsrc;
  u32x4 g0;
  g0[0] = 1u;                                   // count=1, user mode, no gather
  g0[1] = lds_off;                              // lds_addr (bytes)
  g0[2] = (unsigned)(ga & 0xFFFFFFFFull);       // global_addr[31:0]
  g0[3] = (unsigned)((ga >> 32) & 0x1FFFFFFull) | (2u << 30); // addr[56:32], type=2

  i32x8 g1;
  g1[0] = (int)(dsize_code << 16);                           // data_size, wg_mask=0
  g1[1] = (int)((td0 & 0xFFFFu) << 16);                      // tensor_dim0[15:0]
  g1[2] = (int)(((td0 >> 16) & 0xFFFFu) | ((td1 & 0xFFFFu) << 16));
  g1[3] = (int)(((td1 >> 16) & 0xFFFFu) | ((tl0 & 0xFFFFu) << 16));
  g1[4] = (int)(tl1 & 0xFFFFu);                              // tile_dim1; tile_dim2=0
  g1[5] = (int)stride0;                                      // dim0_stride[31:0]
  g1[6] = 0;                                                 // stride0 hi / stride1 lo
  g1[7] = 0;                                                 // stride1 hi

  i32x4 z4 = {0, 0, 0, 0};
#if __clang_major__ >= 23
  i32x8 z8 = {0, 0, 0, 0, 0, 0, 0, 0};
  __builtin_amdgcn_tensor_load_to_lds(g0, g1, z4, z4, z8, 0);
#else
  __builtin_amdgcn_tensor_load_to_lds(g0, g1, z4, z4, 0);
#endif
}
#endif

__device__ __forceinline__ unsigned lds_off_of(const void* p) {
  return (unsigned)(size_t)p;   // low 32 bits of generic LDS address = byte offset
}

// ---- kernel 1: seq_fts{,2} = tanh(seq @ w.T), stored as f16 ----
__global__ void k_feat(const float* __restrict__ seq,
                       const float* __restrict__ w1,
                       const float* __restrict__ w2,
                       _Float16* __restrict__ sf1h,
                       _Float16* __restrict__ sf2h) {
  const int m0 = blockIdx.x * 16;
  const int n0 = blockIdx.y * 16;
  const float* w = blockIdx.z ? w2 : w1;
  _Float16* dst = blockIdx.z ? sf2h : sf1h;

  v8f acc = {};
  for (int kc = 0; kc < F_IN; kc += 32) {
    v16h a = load_A_f32(seq, F_IN, m0, kc);
    v16h b = load_Bt_f32(w, F_IN, n0, kc);
    acc = wmma16(a, b, acc);
  }
  const int lane = threadIdx.x & 31;
  const int n = lane & 15;
  const int mb = (lane >> 4) << 3;
#pragma unroll
  for (int r = 0; r < 8; ++r)
    dst[(size_t)(m0 + mb + r) * F_OUT + n0 + n] = (_Float16)fast_tanh(acc[r]);
}

// ---- kernel 2: flash pass: R = P @ sf1, Spart[mt] = sum(P-rows) ----
// All operands DMA'd into LDS by the TDM; inner loop is ds + wmma only.
__global__ void k_attn(const _Float16* __restrict__ sf1h,
                       const _Float16* __restrict__ sf2h,
                       const float* __restrict__ bias,
                       float* __restrict__ R,
                       float* __restrict__ Spart) {
  __shared__ _Float16 btile[2][32 * F_OUT];  // double-buffered sf1 j-tile (16 KB)
  __shared__ float    bslab[4][16 * 32];     // per-wave bias tile (8 KB)
  __shared__ _Float16 lds_p[4][16 * 32];     // per-wave P staging (4 KB)
  const int wid = threadIdx.x >> 5;
  const int lane = threadIdx.x & 31;
  const int mt = blockIdx.x * 4 + wid;
  const int m0 = mt * 16;
  _Float16* ptile = lds_p[wid];
  float* myslab = bslab[wid];
  const int NT = N_ROWS / 32;

  v16h a2[4];
#pragma unroll
  for (int kc = 0; kc < 4; ++kc) a2[kc] = load_A_f16(sf2h, F_OUT, m0, kc * 32);

  v8f racc[8];
#pragma unroll
  for (int t = 0; t < 8; ++t) racc[t] = (v8f){};

  const float scale = 0.08838834764831845f; // 1/sqrt(128)
  float ssum = 0.f;
  const int n = lane & 15;
  const int mb = (lane >> 4) << 3;

#if HAVE_TDM
  if (wid == 0)   // prologue: btile(0)
    tdm_load_2d(sf1h, lds_off_of(btile[0]), 1, F_OUT, N_ROWS, F_OUT, 32, F_OUT);
#endif

  for (int jt = 0; jt < NT; ++jt) {
    const int j = jt * 32;
    const int cur = jt & 1;
    __syncthreads();  // all waves done reading btile[cur^1]
#if HAVE_TDM
    // per-wave bias 16x32 f32 slab for this j (issued before btile prefetch)
    tdm_load_2d(bias + (size_t)m0 * N_ROWS + j, lds_off_of(myslab), 2,
                (unsigned)(N_ROWS - j), (unsigned)(N_ROWS - m0), 32, 16, N_ROWS);
    if (wid == 0) {
      if (jt + 1 < NT) {
        tdm_load_2d(sf1h + (size_t)(j + 32) * F_OUT, lds_off_of(btile[cur ^ 1]), 1,
                    F_OUT, (unsigned)(N_ROWS - (j + 32)), F_OUT, 32, F_OUT);
        __builtin_amdgcn_s_wait_tensorcnt(1);  // btile(j)+bias(j) done, prefetch in flight
      } else {
        __builtin_amdgcn_s_wait_tensorcnt(0);
      }
    } else {
      __builtin_amdgcn_s_wait_tensorcnt(0);    // my bias slab done
    }
#else
    {
      const uint4* src = (const uint4*)(sf1h + (size_t)j * F_OUT);
      uint4* dst = (uint4*)btile[cur];
#pragma unroll
      for (int i = 0; i < 4; ++i) dst[threadIdx.x + i * 128] = src[threadIdx.x + i * 128];
#pragma unroll
      for (int r = 0; r < 16; r += 2)
        myslab[(r + (lane >> 4)) * 32 + (lane & 15) * 2 + 0] =
            bias[(size_t)(m0 + r + (lane >> 4)) * N_ROWS + j + (lane & 15) * 2 + 0],
        myslab[(r + (lane >> 4)) * 32 + (lane & 15) * 2 + 1] =
            bias[(size_t)(m0 + r + (lane >> 4)) * N_ROWS + j + (lane & 15) * 2 + 1];
    }
#endif
    __syncthreads();  // tiles visible to all waves
    const _Float16* bt = btile[cur];

#pragma unroll
    for (int t = 0; t < 2; ++t) {
      v8f lt = {};
#pragma unroll
      for (int kc = 0; kc < 4; ++kc) {
        v16h b = load_Bt_lds(bt, t * 16, kc * 32);
        lt = wmma16(a2[kc], b, lt);
      }
#pragma unroll
      for (int r = 0; r < 8; ++r) {
        int m = mb + r;
        float p = __expf(lt[r] * scale) * myslab[m * 32 + t * 16 + n];
        ssum += p;
        ptile[m * 32 + t * 16 + n] = (_Float16)p;
      }
    }
    // reshape P tile (C-layout regs) -> A-layout via per-wave-private LDS
    v16h pa = load_A_lds(ptile);
#pragma unroll
    for (int nt = 0; nt < 8; ++nt) {
      v16h b = load_B_rm_lds(bt, nt * 16);
      racc[nt] = wmma16(pa, b, racc[nt]);
    }
  }

#pragma unroll
  for (int nt = 0; nt < 8; ++nt)
#pragma unroll
    for (int r = 0; r < 8; ++r)
      R[(size_t)(m0 + mb + r) * F_OUT + nt * 16 + n] = racc[nt][r];

#pragma unroll
  for (int off = 16; off > 0; off >>= 1) ssum += __shfl_down(ssum, off, 32);
  if (lane == 0) Spart[mt] = ssum;
}

// ---- kernel 3: deterministic tree-reduce of 512 partials ----
__global__ void k_reduce(const float* __restrict__ Spart, float* __restrict__ Ssum) {
  __shared__ float sm[512];
  const int t = threadIdx.x;
  sm[t] = Spart[t];
  __syncthreads();
  for (int s = 256; s > 0; s >>= 1) {
    if (t < s) sm[t] += sm[t + s];
    __syncthreads();
  }
  if (t == 0) Ssum[0] = sm[0];
}

// ---- kernel 4: Rh = f16(R / total) ----
__global__ void k_scale(const float* __restrict__ R, const float* __restrict__ Ssum,
                        _Float16* __restrict__ Rh) {
  const int i = blockIdx.x * blockDim.x + threadIdx.x;
  const float inv = 1.0f / Ssum[0];
  Rh[i] = (_Float16)(R[i] * inv);
}

// ---- kernel 5: out = bias @ Rh, A tiles double-buffered through the TDM ----
__global__ void k_final(const float* __restrict__ bias,
                        const _Float16* __restrict__ Rh,
                        float* __restrict__ out) {
  __shared__ float atile[2][16 * 32];   // 4 KB, single wave per block
  const int m0 = blockIdx.x * 16;
  const int lane = threadIdx.x & 31;
  const int NT = N_ROWS / 32;

  v8f acc[8];
#pragma unroll
  for (int t = 0; t < 8; ++t) acc[t] = (v8f){};

#if HAVE_TDM
  tdm_load_2d(bias + (size_t)m0 * N_ROWS, lds_off_of(atile[0]), 2,
              N_ROWS, (unsigned)(N_ROWS - m0), 32, 16, N_ROWS);
#endif

  for (int jt = 0; jt < NT; ++jt) {
    const int j = jt * 32;
    const int cur = jt & 1;
#if HAVE_TDM
    if (jt + 1 < NT) {
      tdm_load_2d(bias + (size_t)m0 * N_ROWS + j + 32, lds_off_of(atile[cur ^ 1]), 2,
                  (unsigned)(N_ROWS - (j + 32)), (unsigned)(N_ROWS - m0), 32, 16, N_ROWS);
      __builtin_amdgcn_s_wait_tensorcnt(1);
    } else {
      __builtin_amdgcn_s_wait_tensorcnt(0);
    }
#else
    {
      const uint4* src = (const uint4*)(bias + (size_t)m0 * N_ROWS);
      uint4* dst = (uint4*)atile[cur];
      const int ld4 = N_ROWS / 4;
#pragma unroll
      for (int i = 0; i < 4; ++i) {
        int idx = lane + i * 32;                   // 128 uint4 = 16x32 f32
        int r = idx >> 3, c = idx & 7;
        dst[idx] = src[(size_t)r * ld4 + (j >> 2) + c];
      }
    }
#endif
    v16h a = load_A_lds_f32(atile[cur]);
#pragma unroll
    for (int nt = 0; nt < 8; ++nt) {
      v16h b = load_B_rm_f16(Rh, F_OUT, j, nt * 16);
      acc[nt] = wmma16(a, b, acc[nt]);
    }
  }
  const int n = lane & 15;
  const int mb = (lane >> 4) << 3;
#pragma unroll
  for (int nt = 0; nt < 8; ++nt)
#pragma unroll
    for (int r = 0; r < 8; ++r)
      out[(size_t)(m0 + mb + r) * F_OUT + nt * 16 + n] = acc[nt][r];
}

extern "C" void kernel_launch(void* const* d_in, const int* in_sizes, int n_in,
                              void* d_out, int out_size, void* d_ws, size_t ws_size,
                              hipStream_t stream) {
  const float* seq  = (const float*)d_in[0];
  const float* w1   = (const float*)d_in[1];
  const float* w2   = (const float*)d_in[2];
  const float* bias = (const float*)d_in[3];
  float* out = (float*)d_out;

  char* ws = (char*)d_ws;
  const size_t NF = (size_t)N_ROWS * F_OUT;
  _Float16* sf1h = (_Float16*)(ws);                // 2 MiB
  _Float16* sf2h = (_Float16*)(ws + NF * 2);       // 2 MiB
  float*    R    = (float*)   (ws + NF * 4);       // 4 MiB
  _Float16* Rh   = (_Float16*)(ws + NF * 8);       // 2 MiB
  float*    Spart= (float*)   (ws + NF * 10);      // 512 floats
  float*    Ssum = (float*)   (ws + NF * 10 + 4096);

  k_feat  <<<dim3(N_ROWS / 16, F_OUT / 16, 2), 32, 0, stream>>>(seq, w1, w2, sf1h, sf2h);
  k_attn  <<<dim3(N_ROWS / 16 / 4), 128, 0, stream>>>(sf1h, sf2h, bias, R, Spart);
  k_reduce<<<1, 512, 0, stream>>>(Spart, Ssum);
  k_scale <<<dim3((N_ROWS * F_OUT) / 256), 256, 0, stream>>>(R, Ssum, Rh);
  k_final <<<dim3(N_ROWS / 16), 32, 0, stream>>>(bias, Rh, out);
}
